// SelfAttentionBlock_72473278152718
// MI455X (gfx1250) — matrix operand order repaired and checked by reading, not compiled, not run on hardware
//
#include <hip/hip_runtime.h>
#include <hip/hip_bf16.h>
#include <cstdint>
#include <cstddef>

// ---------------------------------------------------------------------------
// CDNA5 (gfx1250) wave32 WMMA types & fragment helpers
// ---------------------------------------------------------------------------
typedef __attribute__((ext_vector_type(16))) _Float16 v16h;
typedef __attribute__((ext_vector_type(8)))  _Float16 v8h;
typedef __attribute__((ext_vector_type(8)))  float    v8f;

union Frag16 { v16h v; v8h h[2]; };

// A-matrix 16x32 f16 layout (ISA 7.12.2): lanes 0-15 hold M=0-15 with
// K = {0..7, 16..23}; lanes 16-31 hold same M rows with K = {8..15, 24..31}.
__device__ __forceinline__ v16h load_frag_a(const _Float16* __restrict__ base,
                                            int ld, int lane) {
  const int m  = lane & 15;
  const int kb = (lane < 16) ? 0 : 8;
  const _Float16* p = base + (size_t)m * ld + kb;
  Frag16 f;
  f.h[0] = *(const v8h*)(p);        // K kb .. kb+7
  f.h[1] = *(const v8h*)(p + 16);   // K kb+16 .. kb+23
  return f.v;
}

// B-matrix 32x16 f16 (KxN), memory holds B^T row-major (row n, K contiguous):
// lanes 0-15 hold N=0-15 with K = {0..7, 16..23}; lanes 16-31 hold K = {8..15, 24..31}.
__device__ __forceinline__ v16h load_frag_b(const _Float16* __restrict__ base,
                                            int ld, int lane) {
  const int n  = lane & 15;
  const int kb = (lane < 16) ? 0 : 8;
  const _Float16* p = base + (size_t)n * ld + kb;
  Frag16 f;
  f.h[0] = *(const v8h*)(p);        // K kb .. kb+7
  f.h[1] = *(const v8h*)(p + 16);   // K kb+16 .. kb+23
  return f.v;
}

__device__ __forceinline__ v8f wmma16(v16h a, v16h b, v8f c) {
  return __builtin_amdgcn_wmma_f32_16x16x32_f16(false, a, false, b,
                                                (short)0, c, false, false);
}

// Async global -> LDS copy (CDNA5, tracked with ASYNCcnt).
// lds_off: byte address within the wave's LDS allocation (low 32 bits of a
// generic pointer to a __shared__ object are the LDS offset on AMDGPU).
__device__ __forceinline__ void async_b128(uint32_t lds_off,
                                           const _Float16* gptr) {
  asm volatile("global_load_async_to_lds_b128 %0, %1, off"
               :: "v"(lds_off), "v"(gptr) : "memory");
}
#define LDSOFF(p) ((uint32_t)(uintptr_t)(p))

// ---------------------------------------------------------------------------
// Problem constants
// ---------------------------------------------------------------------------
#define NB      8
#define SEQ     1024           // 32*32 tokens per batch
#define DM      1024
#define NH      16
#define DH      64
#define CONDD   768
#define TOKENS  (NB * SEQ)     // 8192
#define EPSV    1e-6f

// ---------------------------------------------------------------------------
// Kernel 1: ns = cond @ norm_w^T + 1   -> nsp1[8][1024]
// ---------------------------------------------------------------------------
__global__ __launch_bounds__(256) void ns_kernel(const float* __restrict__ cond,
                                                 const float* __restrict__ nw,
                                                 float* __restrict__ nsp1) {
  const int id = blockIdx.x * 256 + threadIdx.x;  // 8192 outputs
  const int n = id >> 10, d = id & 1023;
  const float* cr = cond + (size_t)n * CONDD;
  const float* wr = nw + (size_t)d * CONDD;
  float acc = 0.f;
  for (int k = 0; k < CONDD; ++k) acc += cr[k] * wr[k];
  nsp1[id] = acc + 1.0f;
}

// ---------------------------------------------------------------------------
// Kernel 2: f32 -> f16 weight conversion
// ---------------------------------------------------------------------------
__global__ __launch_bounds__(256) void cvt_kernel(const float* __restrict__ src,
                                                  _Float16* __restrict__ dst,
                                                  int n) {
  const int i = blockIdx.x * 256 + threadIdx.x;
  if (i < n) dst[i] = (_Float16)src[i];
}

// ---------------------------------------------------------------------------
// Kernel 3: RMS norm, one block per token, writes xn in f16
// ---------------------------------------------------------------------------
__global__ __launch_bounds__(256) void rmsnorm_kernel(const float* __restrict__ x,
                                                      const float* __restrict__ nsp1,
                                                      _Float16* __restrict__ xn) {
  __shared__ float red[256];
  const int t = blockIdx.x;            // token 0..8191
  const int b = t >> 10;
  const int tid = threadIdx.x;
  const float* xr = x + (size_t)t * DM;
  float4 v = *(const float4*)(xr + tid * 4);
  red[tid] = v.x * v.x + v.y * v.y + v.z * v.z + v.w * v.w;
  __syncthreads();
  for (int o = 128; o > 0; o >>= 1) {
    if (tid < o) red[tid] += red[tid + o];
    __syncthreads();
  }
  const float inv = rsqrtf(red[0] * (1.0f / DM) + EPSV);
  const float* nr = nsp1 + (size_t)b * DM;
  _Float16* xo = xn + (size_t)t * DM;
  const float* vv = (const float*)&v;
#pragma unroll
  for (int j = 0; j < 4; ++j) {
    const int c = tid * 4 + j;
    xo[c] = (_Float16)(vv[j] * nr[c] * inv);
  }
}

// ---------------------------------------------------------------------------
// Kernel 4: QKV GEMM (WMMA). C[8192,3072] = xn @ qkv_w^T.
// One wave per 64x64 tile (16 WMMAs per 16 b128 loads -> 4x the reuse of a
// 16x64 tile). Scatters q,k to [b][h][s][64] and v transposed to
// vt[b][h][d][s] so the P@V WMMA gets contiguous B fragments.
// ---------------------------------------------------------------------------
__global__ __launch_bounds__(32) void qkv_gemm(const _Float16* __restrict__ xn,
                                               const _Float16* __restrict__ w,
                                               _Float16* __restrict__ qf,
                                               _Float16* __restrict__ kf,
                                               _Float16* __restrict__ vt) {
  const int lane = threadIdx.x;
  const int m0 = blockIdx.x * 64;
  const int n0 = blockIdx.y * 64;
  v8f acc[4][4] = {};                      // [mt][nt]
  for (int kc = 0; kc < DM; kc += 32) {
    v16h a[4];
#pragma unroll
    for (int mt = 0; mt < 4; ++mt)
      a[mt] = load_frag_a(xn + (size_t)(m0 + mt * 16) * DM + kc, DM, lane);
#pragma unroll
    for (int nt = 0; nt < 4; ++nt) {
      const v16h bfr = load_frag_b(w + (size_t)(n0 + nt * 16) * DM + kc, DM, lane);
#pragma unroll
      for (int mt = 0; mt < 4; ++mt)
        acc[mt][nt] = wmma16(a[mt], bfr, acc[mt][nt]);
    }
  }
  const int hl = lane >> 4, nn = lane & 15;
  const int b = m0 >> 10, s0 = m0 & 1023;
#pragma unroll
  for (int nt = 0; nt < 4; ++nt) {
    const int n = n0 + nt * 16 + nn;
    const int qi = n >> 10;
    const int rem = n & 1023;
    const int h = rem >> 6, d = rem & 63;
#pragma unroll
    for (int mt = 0; mt < 4; ++mt) {
#pragma unroll
      for (int j = 0; j < 8; ++j) {
        const int s = s0 + mt * 16 + j + 8 * hl;
        const _Float16 val = (_Float16)acc[mt][nt][j];
        if (qi == 0)
          qf[(((size_t)(b * NH + h) * SEQ) + s) * DH + d] = val;
        else if (qi == 1)
          kf[(((size_t)(b * NH + h) * SEQ) + s) * DH + d] = val;
        else
          vt[(((size_t)(b * NH + h) * DH) + d) * SEQ + s] = val;
      }
    }
  }
}

// ---------------------------------------------------------------------------
// Kernel 5: per-(b,h,s) L2 norm of q/k (* sqrt(scale)) + RoPE, in place.
// ---------------------------------------------------------------------------
__global__ __launch_bounds__(256) void qk_prep(_Float16* __restrict__ qf,
                                               _Float16* __restrict__ kf,
                                               const float* __restrict__ pos,
                                               const float* __restrict__ scale) {
  const int id = blockIdx.x * 256 + threadIdx.x;   // 8*16*1024 = 131072
  const int b = id >> 14;
  const int h = (id >> 10) & 15;
  const int s = id & 1023;
  const float p0 = pos[((size_t)b * SEQ + s) * 2 + 0];
  const float p1 = pos[((size_t)b * SEQ + s) * 2 + 1];
  const float sq = sqrtf(scale[h]);

  // freqs[h][i] = exp(log(pi) + (i*16 + h) * log(10)/128), i = 0..7
  const float LOGMIN = 1.1447298858494002f;        // ln(pi)
  const float STEP   = 2.302585092994046f / 128.f; // ln(10)/128
  float cs[16], sn[16];
#pragma unroll
  for (int i = 0; i < 8; ++i) {
    const float f = __expf(LOGMIN + (float)(i * 16 + h) * STEP);
    const float th = p0 * f;
    const float tw = p1 * f;
    cs[i] = __cosf(th); sn[i] = __sinf(th);
    cs[i + 8] = __cosf(tw); sn[i + 8] = __sinf(tw);
  }

  for (int pass = 0; pass < 2; ++pass) {
    _Float16* ptr = (pass ? kf : qf) + (size_t)id * DH;
    float v[DH];
    float ssum = 0.f;
#pragma unroll
    for (int d = 0; d < DH; ++d) { v[d] = (float)ptr[d]; ssum += v[d] * v[d]; }
    const float r = sq * rsqrtf(ssum + EPSV);
#pragma unroll
    for (int d = 0; d < DH; ++d) v[d] *= r;
#pragma unroll
    for (int j = 0; j < 16; ++j) {
      const float a = v[j], bb = v[j + 16];
      ptr[j]      = (_Float16)(a * cs[j] - bb * sn[j]);
      ptr[j + 16] = (_Float16)(bb * cs[j] + a * sn[j]);
    }
#pragma unroll
    for (int d = 32; d < DH; ++d) ptr[d] = (_Float16)v[d];
  }
}

// ---------------------------------------------------------------------------
// Kernel 6: flash attention. One wave per (b, h, 16-query tile).
// K/V tiles are streamed into LDS with global_load_async_to_lds_b128,
// double-buffered across 32-key chunks; in-order ASYNCcnt completion means
// "wait <= 16" guarantees the current buffer is resident while the next
// streams. Score/PV matmuls are WMMA fed from LDS (ds_load_b128).
// ---------------------------------------------------------------------------
__global__ __launch_bounds__(32) void attn_kernel(const _Float16* __restrict__ qf,
                                                  const _Float16* __restrict__ kf,
                                                  const _Float16* __restrict__ vt,
                                                  _Float16* __restrict__ of) {
  __shared__ __align__(16) _Float16 ktile[2][32 * 64];  // [key][d]
  __shared__ __align__(16) _Float16 vtile[2][64 * 32];  // [d][key]
  __shared__ __align__(16) _Float16 pp[16 * 32];
  const int lane = threadIdx.x;
  const int qt = blockIdx.x, h = blockIdx.y, b = blockIdx.z;
  const _Float16* qbase = qf + ((size_t)(b * NH + h) * SEQ) * DH;
  const _Float16* kbase = kf + ((size_t)(b * NH + h) * SEQ) * DH;
  const _Float16* vbase = vt + ((size_t)(b * NH + h) * DH) * SEQ;

  const v16h aq0 = load_frag_a(qbase + (size_t)qt * 16 * DH, DH, lane);
  const v16h aq1 = load_frag_a(qbase + (size_t)qt * 16 * DH + 32, DH, lane);

  // Stage chunk 0 into buffer 0. K tile: 4KB contiguous (32 keys x 128B).
  // V tile: 64 d-rows x 64B segments out of the [d][SEQ] transposed layout.
  {
    const uint32_t lk = LDSOFF(&ktile[0][0]);
    const uint32_t lv = LDSOFF(&vtile[0][0]);
#pragma unroll
    for (int i = 0; i < 8; ++i) {
      const int c = i * 32 + lane;
      async_b128(lk + c * 16, kbase + c * 8);
    }
#pragma unroll
    for (int i = 0; i < 8; ++i) {
      const int c = i * 32 + lane;
      async_b128(lv + c * 16, vbase + (c >> 2) * SEQ + (c & 3) * 8);
    }
  }

  v8f co0 = {}, co1 = {}, co2 = {}, co3 = {};
  float mrow[8], lrow[8];
#pragma unroll
  for (int j = 0; j < 8; ++j) { mrow[j] = -1e30f; lrow[j] = 0.f; }
  const int hl = lane >> 4, nn = lane & 15;

  for (int kc = 0; kc < SEQ / 32; ++kc) {
    const int buf = kc & 1;
    if (kc + 1 < SEQ / 32) {
      const _Float16* knext = kbase + (size_t)(kc + 1) * 32 * DH;
      const _Float16* vnext = vbase + (size_t)(kc + 1) * 32;
      const uint32_t lk = LDSOFF(&ktile[buf ^ 1][0]);
      const uint32_t lv = LDSOFF(&vtile[buf ^ 1][0]);
#pragma unroll
      for (int i = 0; i < 8; ++i) {
        const int c = i * 32 + lane;
        async_b128(lk + c * 16, knext + c * 8);
      }
#pragma unroll
      for (int i = 0; i < 8; ++i) {
        const int c = i * 32 + lane;
        async_b128(lv + c * 16, vnext + (c >> 2) * SEQ + (c & 3) * 8);
      }
      asm volatile("s_wait_asynccnt 0x10" ::: "memory");  // current buf done
    } else {
      asm volatile("s_wait_asynccnt 0x0" ::: "memory");
    }

    const _Float16* kp = &ktile[buf][0];
    v8f s0 = {}, s1 = {};
    s0 = wmma16(aq0, load_frag_b(kp, DH, lane), s0);                 // keys 0-15, d 0-31
    s0 = wmma16(aq1, load_frag_b(kp + 32, DH, lane), s0);            // keys 0-15, d 32-63
    s1 = wmma16(aq0, load_frag_b(kp + 16 * DH, DH, lane), s1);       // keys 16-31
    s1 = wmma16(aq1, load_frag_b(kp + 16 * DH + 32, DH, lane), s1);

#pragma unroll
    for (int j = 0; j < 8; ++j) {
      float mx = fmaxf(s0[j], s1[j]);
      mx = fmaxf(mx, __shfl_xor(mx, 1, 32));
      mx = fmaxf(mx, __shfl_xor(mx, 2, 32));
      mx = fmaxf(mx, __shfl_xor(mx, 4, 32));
      mx = fmaxf(mx, __shfl_xor(mx, 8, 32));
      const float mnew = fmaxf(mrow[j], mx);
      const float sf = __expf(mrow[j] - mnew);
      const float p0 = __expf(s0[j] - mnew);
      const float p1 = __expf(s1[j] - mnew);
      float rs = p0 + p1;
      rs += __shfl_xor(rs, 1, 32);
      rs += __shfl_xor(rs, 2, 32);
      rs += __shfl_xor(rs, 4, 32);
      rs += __shfl_xor(rs, 8, 32);
      lrow[j] = lrow[j] * sf + rs;
      mrow[j] = mnew;
      const int row = j + 8 * hl;
      pp[row * 32 + nn]      = (_Float16)p0;
      pp[row * 32 + 16 + nn] = (_Float16)p1;
      co0[j] *= sf; co1[j] *= sf; co2[j] *= sf; co3[j] *= sf;
    }
    __syncthreads();

    v16h ap;
    {
      const int m = lane & 15;
      const int kb = (lane < 16) ? 0 : 8;
      const _Float16* p = &pp[m * 32 + kb];
      Frag16 f;
      f.h[0] = *(const v8h*)(p);
      f.h[1] = *(const v8h*)(p + 16);
      ap = f.v;
    }
    const _Float16* vp = &vtile[buf][0];
    co0 = wmma16(ap, load_frag_b(vp + 0 * 16 * 32, 32, lane), co0);
    co1 = wmma16(ap, load_frag_b(vp + 1 * 16 * 32, 32, lane), co1);
    co2 = wmma16(ap, load_frag_b(vp + 2 * 16 * 32, 32, lane), co2);
    co3 = wmma16(ap, load_frag_b(vp + 3 * 16 * 32, 32, lane), co3);
    __syncthreads();
  }

#pragma unroll
  for (int j = 0; j < 8; ++j) {
    const int row = j + 8 * hl;
    const float inv = 1.0f / lrow[j];
    const size_t srow = (size_t)b * SEQ + (size_t)qt * 16 + row;
    _Float16* orow = of + srow * DM + h * DH;
    orow[ 0 + nn] = (_Float16)(co0[j] * inv);
    orow[16 + nn] = (_Float16)(co1[j] * inv);
    orow[32 + nn] = (_Float16)(co2[j] * inv);
    orow[48 + nn] = (_Float16)(co3[j] * inv);
  }
}

// ---------------------------------------------------------------------------
// Kernel 7: output GEMM (WMMA) + skip. out[8192,1024] = o @ out_w^T + x.
// One wave per 64x64 tile.
// ---------------------------------------------------------------------------
__global__ __launch_bounds__(32) void out_gemm(const _Float16* __restrict__ of,
                                               const _Float16* __restrict__ w,
                                               const float* __restrict__ x,
                                               float* __restrict__ out) {
  const int lane = threadIdx.x;
  const int m0 = blockIdx.x * 64;
  const int n0 = blockIdx.y * 64;
  v8f acc[4][4] = {};                      // [mt][nt]
  for (int kc = 0; kc < DM; kc += 32) {
    v16h a[4];
#pragma unroll
    for (int mt = 0; mt < 4; ++mt)
      a[mt] = load_frag_a(of + (size_t)(m0 + mt * 16) * DM + kc, DM, lane);
#pragma unroll
    for (int nt = 0; nt < 4; ++nt) {
      const v16h bfr = load_frag_b(w + (size_t)(n0 + nt * 16) * DM + kc, DM, lane);
#pragma unroll
      for (int mt = 0; mt < 4; ++mt)
        acc[mt][nt] = wmma16(a[mt], bfr, acc[mt][nt]);
    }
  }
  const int hl = lane >> 4, nn = lane & 15;
#pragma unroll
  for (int mt = 0; mt < 4; ++mt) {
#pragma unroll
    for (int j = 0; j < 8; ++j) {
      const size_t m = (size_t)m0 + mt * 16 + j + 8 * hl;
      const float* xr = x + m * DM + n0;
      float* orr = out + m * DM + n0;
      orr[ 0 + nn] = acc[mt][0][j] + xr[ 0 + nn];
      orr[16 + nn] = acc[mt][1][j] + xr[16 + nn];
      orr[32 + nn] = acc[mt][2][j] + xr[32 + nn];
      orr[48 + nn] = acc[mt][3][j] + xr[48 + nn];
    }
  }
}

// ---------------------------------------------------------------------------
// Workspace layout (bytes, 256-aligned); total ~88 MB
// ---------------------------------------------------------------------------
static constexpr size_t OFF_NSP1 = 0;                                   // 32 KB
static constexpr size_t OFF_XN   = OFF_NSP1 + (size_t)NB * DM * 4;      // 16 MB f16
static constexpr size_t OFF_QKVW = OFF_XN   + (size_t)TOKENS * DM * 2;  // 6 MB f16
static constexpr size_t OFF_OUTW = OFF_QKVW + (size_t)3 * DM * DM * 2;  // 2 MB f16
static constexpr size_t OFF_Q    = OFF_OUTW + (size_t)DM * DM * 2;      // 16 MB f16
static constexpr size_t OFF_K    = OFF_Q    + (size_t)NB * NH * SEQ * DH * 2;
static constexpr size_t OFF_VT   = OFF_K    + (size_t)NB * NH * SEQ * DH * 2;
static constexpr size_t OFF_O    = OFF_VT   + (size_t)NB * NH * SEQ * DH * 2;

extern "C" void kernel_launch(void* const* d_in, const int* in_sizes, int n_in,
                              void* d_out, int out_size, void* d_ws, size_t ws_size,
                              hipStream_t stream) {
  (void)in_sizes; (void)n_in; (void)out_size; (void)ws_size;
  const float* x      = (const float*)d_in[0];  // [8,32,32,1024]
  const float* pos    = (const float*)d_in[1];  // [8,32,32,2]
  const float* cond   = (const float*)d_in[2];  // [8,768]
  const float* norm_w = (const float*)d_in[3];  // [1024,768]
  const float* qkv_w  = (const float*)d_in[4];  // [3072,1024]
  const float* out_w  = (const float*)d_in[5];  // [1024,1024]
  const float* scale  = (const float*)d_in[6];  // [16]
  float* out = (float*)d_out;

  char* ws = (char*)d_ws;
  float*    nsp1  = (float*)   (ws + OFF_NSP1);
  _Float16* xn    = (_Float16*)(ws + OFF_XN);
  _Float16* qkvwh = (_Float16*)(ws + OFF_QKVW);
  _Float16* outwh = (_Float16*)(ws + OFF_OUTW);
  _Float16* qfp   = (_Float16*)(ws + OFF_Q);
  _Float16* kfp   = (_Float16*)(ws + OFF_K);
  _Float16* vtp   = (_Float16*)(ws + OFF_VT);
  _Float16* ofp   = (_Float16*)(ws + OFF_O);

  // weights -> f16
  cvt_kernel<<<(3 * DM * DM + 255) / 256, 256, 0, stream>>>(qkv_w, qkvwh, 3 * DM * DM);
  cvt_kernel<<<(DM * DM + 255) / 256, 256, 0, stream>>>(out_w, outwh, DM * DM);
  // ns + 1
  ns_kernel<<<(NB * DM) / 256, 256, 0, stream>>>(cond, norm_w, nsp1);
  // RMS norm -> xn (f16)
  rmsnorm_kernel<<<TOKENS, 256, 0, stream>>>(x, nsp1, xn);
  // QKV projection (WMMA), scatter q/k/[v transposed]
  {
    dim3 g(TOKENS / 64, (3 * DM) / 64);
    qkv_gemm<<<g, 32, 0, stream>>>(xn, qkvwh, qfp, kfp, vtp);
  }
  // q/k L2 norm + RoPE
  qk_prep<<<(NB * NH * SEQ) / 256, 256, 0, stream>>>(qfp, kfp, pos, scale);
  // attention (WMMA flash + async LDS staging)
  {
    dim3 g(SEQ / 16, NH, NB);
    attn_kernel<<<g, 32, 0, stream>>>(qfp, kfp, vtp, ofp);
  }
  // output projection + skip (WMMA)
  {
    dim3 g(TOKENS / 64, DM / 64);
    out_gemm<<<g, 32, 0, stream>>>(ofp, outwh, x, out);
  }
}
